// PALM_45767171506362
// MI455X (gfx1250) — matrix operand order repaired and checked by reading, not compiled, not run on hardware
//
#include <hip/hip_runtime.h>

// ---------------- types ----------------
typedef __bf16 bf16;
typedef __attribute__((ext_vector_type(16))) __bf16 v16bf;
typedef __attribute__((ext_vector_type(8)))  __bf16 v8bf;
typedef __attribute__((ext_vector_type(8)))  float  v8f;

#define NN 4096      // N = V*B
#define BB 2048
#define DD 512
#define KK 8192      // NUM_CLASSES*CACHE
#define CC 128
#define MAXR 1024
#define INV_EPS 20.0f   // 1/0.05
#define INV_TEMP 10.0f  // 1/0.1

__device__ __forceinline__ v16bf join16(v8bf a, v8bf b) {
  return __builtin_shufflevector(a, b, 0,1,2,3,4,5,6,7,8,9,10,11,12,13,14,15);
}

// fragment loaders (layouts per CDNA5 ISA 7.12.2; both in table K order)
__device__ __forceinline__ void loadA4(const bf16* __restrict__ arow, int Kd, int ka,
                                       v16bf* dst) {
#pragma unroll
  for (int mi = 0; mi < 4; ++mi) {
    const bf16* p = arow + (size_t)mi * 16 * Kd + ka;
    dst[mi] = join16(*(const v8bf*)p, *(const v8bf*)(p + 16));
  }
}
__device__ __forceinline__ void loadB2(const bf16* __restrict__ brow, int Kd, int kb,
                                       v16bf* dst) {
#pragma unroll
  for (int ni = 0; ni < 2; ++ni) {
    const bf16* p = brow + (size_t)ni * 16 * Kd + kb;
    dst[ni] = join16(*(const v8bf*)p, *(const v8bf*)(p + 16));
  }
}

// ---------------- init / small utility kernels ----------------
__global__ void palm_init(float* colacc, float* colsum, int* cls_count,
                          float* acc, float* u, float* v) {
  int i = blockIdx.x * blockDim.x + threadIdx.x;
  if (i < KK) { colacc[i] = 0.f; colsum[i] = 0.f; u[i] = 1.f; }
  if (i < NN) v[i] = 1.f;
  if (i < CC) cls_count[i] = 0;
  if (i < 2)  acc[i] = 0.f;
}

__global__ void palm_reinit_uv(float* u, float* v) {
  int i = blockIdx.x * blockDim.x + threadIdx.x;
  if (i < KK) u[i] = 1.f;
  if (i < NN) v[i] = 1.f;
}

// features [2048,2,512] f32 -> feats bf16 [4096,512] view-major
__global__ void palm_cvt_feats(const float* __restrict__ features, bf16* __restrict__ fB) {
  int idx = blockIdx.x * blockDim.x + threadIdx.x;   // over NN*DD
  int n = idx >> 9, d = idx & 511;
  int b = n & (BB - 1), vv = n >> 11;                // view 0 or 1
  fB[idx] = (bf16)features[(size_t)b * 1024 + vv * 512 + d];
}

__global__ void palm_cvt_protos(const float* __restrict__ protos, bf16* __restrict__ pB) {
  int idx = blockIdx.x * blockDim.x + threadIdx.x;   // over KK*DD
  pB[idx] = (bf16)protos[idx];
}

__global__ void palm_classlist(const int* __restrict__ targets, int* cls_count, int* cls_rows) {
  int n = blockIdx.x * blockDim.x + threadIdx.x;
  if (n >= NN) return;
  int c = targets[n & (BB - 1)];
  int pos = atomicAdd(&cls_count[c], 1);
  if (pos < MAXR) cls_rows[c * MAXR + pos] = n;
}

// ---------------- WMMA GEMM: C[M,K] = A[M,D] @ B[K,D]^T (bf16 in, f32 out) ------
// block = 8 waves; block tile 64(M) x 256(cols); wave tile 64x32 (4x2 WMMA frags)
// register double-buffered over K so loads for k+32 overlap WMMAs on k.
__global__ __launch_bounds__(256) void palm_gemm(const bf16* __restrict__ A,
                                                 const bf16* __restrict__ B,
                                                 float* __restrict__ C,
                                                 int M, int Ncols, int Kd) {
  const int lane = threadIdx.x & 31;
  const int wid  = threadIdx.x >> 5;
  const int m0 = blockIdx.x * 64;
  const int n0 = blockIdx.y * 256 + wid * 32;
  const int lm = lane & 15;
  const int hi = lane >> 4;
  const int aoff = hi ? 8 : 0;     // A frag K sub-offset
  const int boff = hi ? 8 : 0;     // B frag K sub-offset

  v8f acc[4][2] = {};
  const bf16* arow = A + (size_t)(m0 + lm) * Kd;
  const bf16* brow = B + (size_t)(n0 + lm) * Kd;

  v16bf a0[4], b0[2], a1[4], b1[2];
  loadA4(arow, Kd, aoff, a0);
  loadB2(brow, Kd, boff, b0);

  for (int k0 = 0; k0 < Kd; k0 += 64) {
    // issue loads for k0+32 before consuming k0
    loadA4(arow, Kd, k0 + 32 + aoff, a1);
    loadB2(brow, Kd, k0 + 32 + boff, b1);
#pragma unroll
    for (int mi = 0; mi < 4; ++mi)
#pragma unroll
      for (int ni = 0; ni < 2; ++ni)
        acc[mi][ni] = __builtin_amdgcn_wmma_f32_16x16x32_bf16(
            false, a0[mi], false, b0[ni], (short)0, acc[mi][ni], false, false);
    // issue loads for k0+64 before consuming k0+32
    if (k0 + 64 < Kd) {
      loadA4(arow, Kd, k0 + 64 + aoff, a0);
      loadB2(brow, Kd, k0 + 64 + boff, b0);
    }
#pragma unroll
    for (int mi = 0; mi < 4; ++mi)
#pragma unroll
      for (int ni = 0; ni < 2; ++ni)
        acc[mi][ni] = __builtin_amdgcn_wmma_f32_16x16x32_bf16(
            false, a1[mi], false, b1[ni], (short)0, acc[mi][ni], false, false);
  }
#pragma unroll
  for (int mi = 0; mi < 4; ++mi)
#pragma unroll
    for (int ni = 0; ni < 2; ++ni) {
      int col = n0 + ni * 16 + lm;
      int rbase = m0 + mi * 16 + (hi ? 8 : 0);
#pragma unroll
      for (int j = 0; j < 8; ++j)
        C[(size_t)(rbase + j) * Ncols + col] = acc[mi][ni][j];
    }
}

// ---------------- row max over S ----------------
__global__ __launch_bounds__(256) void palm_rowmax(const float* __restrict__ S, float* rm) {
  __shared__ float red[256];
  int n = blockIdx.x;
  float m = -1e30f;
  for (int k = threadIdx.x; k < KK; k += 256) m = fmaxf(m, S[(size_t)n * KK + k]);
  red[threadIdx.x] = m; __syncthreads();
  for (int o = 128; o >= 1; o >>= 1) {
    if (threadIdx.x < o) red[threadIdx.x] = fmaxf(red[threadIdx.x], red[threadIdx.x + o]);
    __syncthreads();
  }
  if (threadIdx.x == 0) rm[n] = red[0];
}

// ---------------- sinkhorn passes (scaling-vector form) ----------------
__global__ __launch_bounds__(256) void palm_sk_col(const float* __restrict__ S,
                                                   const float* __restrict__ rm,
                                                   const float* __restrict__ v,
                                                   float* __restrict__ colacc) {
  int k = blockIdx.x * 256 + threadIdx.x;
  int nBeg = blockIdx.y * 512, nEnd = nBeg + 512;
  float sum = 0.f;
  for (int n = nBeg; n < nEnd; ++n)
    sum += __expf((S[(size_t)n * KK + k] - rm[n]) * INV_EPS) * v[n];
  atomicAdd(&colacc[k], sum);
}

__global__ void palm_sk_ufin(float* u, float* colacc) {
  int k = blockIdx.x * blockDim.x + threadIdx.x;
  if (k < KK) { u[k] = 1.0f / ((float)KK * fmaxf(colacc[k], 1e-12f)); colacc[k] = 0.f; }
}

__global__ __launch_bounds__(256) void palm_sk_row(const float* __restrict__ S,
                                                   const float* __restrict__ rm,
                                                   const float* __restrict__ u,
                                                   float* __restrict__ v) {
  __shared__ float red[256];
  int n = blockIdx.x;
  float r = rm[n], sum = 0.f;
  for (int k = threadIdx.x; k < KK; k += 256)
    sum += __expf((S[(size_t)n * KK + k] - r) * INV_EPS) * u[k];
  red[threadIdx.x] = sum; __syncthreads();
  for (int o = 128; o >= 1; o >>= 1) {
    if (threadIdx.x < o) red[threadIdx.x] += red[threadIdx.x + o];
    __syncthreads();
  }
  if (threadIdx.x == 0) v[n] = 1.0f / ((float)NN * fmaxf(red[0], 1e-12f));
}

// ---------------- top-10 of the 64 class-cache slots, row-l1-norm, colsum ------
__global__ void palm_topk(const float* __restrict__ S, const float* __restrict__ rm,
                          const float* __restrict__ u, const float* __restrict__ v,
                          const int* __restrict__ targets, float* __restrict__ um_r,
                          unsigned long long* __restrict__ tmask, float* colsum) {
  int n = blockIdx.x * blockDim.x + threadIdx.x;
  if (n >= NN) return;
  int c = targets[n & (BB - 1)];
  float r = rm[n], vn = v[n];
  float q[64];
  for (int j = 0; j < 64; ++j) {
    int k = c + (j << 7);
    q[j] = __expf((S[(size_t)n * KK + k] - r) * INV_EPS) * u[k] * vn * (float)NN;
  }
  unsigned long long msk = 0ull; float rsum = 0.f;
  for (int t = 0; t < 10; ++t) {           // K_TOP = 10
    int bj = 0; float bv = -1e30f;
    for (int j = 0; j < 64; ++j)
      if (!((msk >> j) & 1ull) && q[j] > bv) { bv = q[j]; bj = j; }
    msk |= (1ull << bj); rsum += bv;
  }
  float inv = 1.0f / fmaxf(rsum, 1e-12f);
  tmask[n] = msk;
  for (int j = 0; j < 64; ++j) {
    float w = ((msk >> j) & 1ull) ? q[j] * inv : 0.f;
    um_r[n * 64 + j] = w;
    if (w != 0.f) atomicAdd(&colsum[c + (j << 7)], w);
  }
}

// ---------------- upd[k,:] = sum_n um_final[n,k] * feats[n,:] (sparse) ----------
__global__ __launch_bounds__(256) void palm_upd(const float* __restrict__ features,
                                                const int* __restrict__ cls_count,
                                                const int* __restrict__ cls_rows,
                                                const float* __restrict__ um_r,
                                                const float* __restrict__ colsum,
                                                float* __restrict__ upd) {
  int k = blockIdx.x;
  int c = k & (CC - 1), j = k >> 7;
  float inv = 1.0f / fmaxf(colsum[k], 1e-12f);
  int cnt = cls_count[c]; if (cnt > MAXR) cnt = MAXR;
  int d0 = threadIdx.x, d1 = threadIdx.x + 256;
  float a0 = 0.f, a1 = 0.f;
  for (int i = 0; i < cnt; ++i) {
    int n = cls_rows[c * MAXR + i];
    float w = um_r[n * 64 + j];
    if (w != 0.f) {
      w *= inv;
      const float* f = features + ((n < BB) ? ((size_t)n * 1024)
                                            : ((size_t)(n - BB) * 1024 + 512));
      a0 += w * f[d0]; a1 += w * f[d1];
    }
  }
  upd[(size_t)k * DD + d0] = a0;
  upd[(size_t)k * DD + d1] = a1;
}

// ---------------- protos_new = l2n(0.99*protos + 0.01*upd) -> bf16 -------------
__global__ __launch_bounds__(256) void palm_pnew(const float* __restrict__ protos,
                                                 const float* __restrict__ upd,
                                                 bf16* __restrict__ pnewB) {
  __shared__ float red[256];
  int k = blockIdx.x;
  size_t base = (size_t)k * DD;
  float v0 = 0.99f * protos[base + threadIdx.x]       + 0.01f * upd[base + threadIdx.x];
  float v1 = 0.99f * protos[base + 256 + threadIdx.x] + 0.01f * upd[base + 256 + threadIdx.x];
  red[threadIdx.x] = v0 * v0 + v1 * v1; __syncthreads();
  for (int o = 128; o >= 1; o >>= 1) {
    if (threadIdx.x < o) red[threadIdx.x] += red[threadIdx.x + o];
    __syncthreads();
  }
  float inv = 1.0f / fmaxf(sqrtf(red[0]), 1e-12f);
  pnewB[base + threadIdx.x]       = (bf16)(v0 * inv);
  pnewB[base + 256 + threadIdx.x] = (bf16)(v1 * inv);
}

// ---------------- neg[n] = logsumexp_k(S2/T); acc[0] += neg --------------------
__global__ __launch_bounds__(256) void palm_neg(const float* __restrict__ S,
                                                const float* __restrict__ rm, float* acc) {
  __shared__ float red[256];
  int n = blockIdx.x;
  float r = rm[n], sum = 0.f;
  for (int k = threadIdx.x; k < KK; k += 256)
    sum += __expf((S[(size_t)n * KK + k] - r) * INV_TEMP);
  red[threadIdx.x] = sum; __syncthreads();
  for (int o = 128; o >= 1; o >>= 1) {
    if (threadIdx.x < o) red[threadIdx.x] += red[threadIdx.x + o];
    __syncthreads();
  }
  if (threadIdx.x == 0) atomicAdd(&acc[0], r * INV_TEMP + logf(red[0]));
}

// ---------------- pos[n] via masked, row-normalized Q2; acc[0] -= pos ----------
__global__ void palm_pos(const float* __restrict__ S, const float* __restrict__ rm,
                         const float* __restrict__ u, const float* __restrict__ v,
                         const int* __restrict__ targets,
                         const unsigned long long* __restrict__ tmask, float* acc) {
  int n = blockIdx.x * blockDim.x + threadIdx.x;
  if (n >= NN) return;
  int c = targets[n & (BB - 1)];
  unsigned long long msk = tmask[n];
  float r = rm[n], vn = v[n];
  float swq = 0.f, swql = 0.f;
  for (int j = 0; j < 64; ++j)
    if ((msk >> j) & 1ull) {
      int k = c + (j << 7);
      float sv = S[(size_t)n * KK + k];
      float q = __expf((sv - r) * INV_EPS) * u[k] * vn * (float)NN;
      swq += q; swql += q * sv * INV_TEMP;
    }
  atomicAdd(&acc[0], -(swql / fmaxf(swq, 1e-12f)));
}

// ---------------- proto-contra: fused online-softmax gram via WMMA -------------
// one wave per 16-row tile of p; streams all 512 column tiles; the K-loop is
// register double-buffered so fragment loads overlap the WMMA chain.
__global__ __launch_bounds__(256) void palm_pcon(const bf16* __restrict__ P, float* acc) {
  const int lane = threadIdx.x & 31;
  const int wid  = threadIdx.x >> 5;
  const int m0 = (blockIdx.x * 8 + wid) * 16;
  const int lm = lane & 15;
  const int hi = lane >> 4;
  const int aoff = hi ? 8 : 0;
  const int boff = hi ? 8 : 0;

  float m[8], s[8], ss[8];
#pragma unroll
  for (int j = 0; j < 8; ++j) { m[j] = -1e30f; s[j] = 0.f; ss[j] = 0.f; }

  const bf16* arow = P + (size_t)(m0 + lm) * DD;
  for (int n0 = 0; n0 < KK; n0 += 16) {
    const bf16* brow = P + (size_t)(n0 + lm) * DD;
    v8f c = {};
    v16bf a0, b0, a1, b1;
    {
      const bf16* pa = arow + aoff;
      a0 = join16(*(const v8bf*)pa, *(const v8bf*)(pa + 16));
      const bf16* pb = brow + boff;
      b0 = join16(*(const v8bf*)pb, *(const v8bf*)(pb + 16));
    }
    for (int k0 = 0; k0 < DD; k0 += 64) {
      const bf16* pa1 = arow + k0 + 32 + aoff;
      a1 = join16(*(const v8bf*)pa1, *(const v8bf*)(pa1 + 16));
      const bf16* pb1 = brow + k0 + 32 + boff;
      b1 = join16(*(const v8bf*)pb1, *(const v8bf*)(pb1 + 16));
      c = __builtin_amdgcn_wmma_f32_16x16x32_bf16(false, a0, false, b0, (short)0, c,
                                                  false, false);
      if (k0 + 64 < DD) {
        const bf16* pa0 = arow + k0 + 64 + aoff;
        a0 = join16(*(const v8bf*)pa0, *(const v8bf*)(pa0 + 16));
        const bf16* pb0 = brow + k0 + 64 + boff;
        b0 = join16(*(const v8bf*)pb0, *(const v8bf*)(pb0 + 16));
      }
      c = __builtin_amdgcn_wmma_f32_16x16x32_bf16(false, a1, false, b1, (short)0, c,
                                                  false, false);
    }
    const int col = n0 + lm;
#pragma unroll
    for (int j = 0; j < 8; ++j) {
      int row = m0 + j + (hi ? 8 : 0);
      float val = c[j] * 2.0f;                       // adc = dot / 0.5
      bool diag = (row == col);
      bool same = (((row ^ col) & (CC - 1)) == 0) && !diag;
      float mc = val;
#pragma unroll
      for (int o = 8; o >= 1; o >>= 1) mc = fmaxf(mc, __shfl_xor(mc, o, 16));
      float mn = fmaxf(m[j], mc);
      float e  = diag ? 0.f : __expf(val - mn);      // lmask kills diagonal in exp-sum
      float sv = same ? val : 0.f;                   // pmask*lmask same-class sum
#pragma unroll
      for (int o = 8; o >= 1; o >>= 1) {
        e  += __shfl_xor(e,  o, 16);
        sv += __shfl_xor(sv, o, 16);
      }
      s[j] = s[j] * __expf(m[j] - mn) + e;
      ss[j] += sv;
      m[j] = mn;
    }
  }
  if (lm == 0) {                                     // lanes 0 and 16: 8 rows each
    float t = 0.f;
#pragma unroll
    for (int j = 0; j < 8; ++j)                      // neg_p - pos_p per row
      t += logf(s[j]) - ss[j] * (1.0f / 63.0f) + m[j];
    atomicAdd(&acc[1], t);
  }
}

// ---------------- final scalar -------------------------------------------------
__global__ void palm_fin(const float* acc, float* out) {
  out[0] = acc[0] * (1.0f / (float)NN) + acc[1] * (1.0f / (float)KK);
}

// ---------------- host orchestration ------------------------------------------
extern "C" void kernel_launch(void* const* d_in, const int* in_sizes, int n_in,
                              void* d_out, int out_size, void* d_ws, size_t ws_size,
                              hipStream_t stream) {
  const float* features = (const float*)d_in[0];
  const float* protos   = (const float*)d_in[1];
  const int*   targets  = (const int*)d_in[2];

  char* w = (char*)d_ws;
  size_t o = 0;
  auto alloc = [&](size_t bytes) -> void* {
    void* p = w + o;
    o = (o + bytes + 255) & ~(size_t)255;
    return p;
  };
  float* S       = (float*)alloc((size_t)NN * KK * 4);   // 128 MB, reused for S1 then S2
  bf16*  featsB  = (bf16*) alloc((size_t)NN * DD * 2);
  bf16*  protosB = (bf16*) alloc((size_t)KK * DD * 2);
  bf16*  pnewB   = (bf16*) alloc((size_t)KK * DD * 2);
  float* upd     = (float*)alloc((size_t)KK * DD * 4);
  float* um_r    = (float*)alloc((size_t)NN * 64 * 4);
  unsigned long long* tmask = (unsigned long long*)alloc((size_t)NN * 8);
  float* rowmax  = (float*)alloc((size_t)NN * 4);
  float* u       = (float*)alloc((size_t)KK * 4);
  float* v       = (float*)alloc((size_t)NN * 4);
  float* colacc  = (float*)alloc((size_t)KK * 4);
  float* colsum  = (float*)alloc((size_t)KK * 4);
  int*   cls_count = (int*)alloc(CC * 4);
  int*   cls_rows  = (int*)alloc((size_t)CC * MAXR * 4);
  float* acc     = (float*)alloc(2 * 4);

  // init + conversions + class lists
  palm_init<<<KK / 256, 256, 0, stream>>>(colacc, colsum, cls_count, acc, u, v);
  palm_cvt_feats <<<(NN * DD) / 256, 256, 0, stream>>>(features, featsB);
  palm_cvt_protos<<<(KK * DD) / 256, 256, 0, stream>>>(protos, protosB);
  palm_classlist <<<NN / 256, 256, 0, stream>>>(targets, cls_count, cls_rows);

  // S1 = feats @ protos^T ; sinkhorn 1
  palm_gemm<<<dim3(NN / 64, KK / 256), 256, 0, stream>>>(featsB, protosB, S, NN, KK, DD);
  palm_rowmax<<<NN, 256, 0, stream>>>(S, rowmax);
  for (int it = 0; it < 3; ++it) {
    palm_sk_col<<<dim3(KK / 256, NN / 512), 256, 0, stream>>>(S, rowmax, v, colacc);
    palm_sk_ufin<<<KK / 256, 256, 0, stream>>>(u, colacc);
    palm_sk_row<<<NN, 256, 0, stream>>>(S, rowmax, u, v);
  }

  // um (top-10 over 64 class slots), EMA prototype update
  palm_topk<<<NN / 256, 256, 0, stream>>>(S, rowmax, u, v, targets, um_r, tmask, colsum);
  palm_upd <<<KK, 256, 0, stream>>>(features, cls_count, cls_rows, um_r, colsum, upd);
  palm_pnew<<<KK, 256, 0, stream>>>(protos, upd, pnewB);

  // S2 = feats @ protos_new^T ; sinkhorn 2
  palm_gemm<<<dim3(NN / 64, KK / 256), 256, 0, stream>>>(featsB, pnewB, S, NN, KK, DD);
  palm_rowmax<<<NN, 256, 0, stream>>>(S, rowmax);
  palm_reinit_uv<<<KK / 256, 256, 0, stream>>>(u, v);
  for (int it = 0; it < 3; ++it) {
    palm_sk_col<<<dim3(KK / 256, NN / 512), 256, 0, stream>>>(S, rowmax, v, colacc);
    palm_sk_ufin<<<KK / 256, 256, 0, stream>>>(u, colacc);
    palm_sk_row<<<NN, 256, 0, stream>>>(S, rowmax, u, v);
  }

  // MLE loss terms and proto-contra
  palm_neg<<<NN, 256, 0, stream>>>(S, rowmax, acc);
  palm_pos<<<NN / 256, 256, 0, stream>>>(S, rowmax, u, v, targets, tmask, acc);
  palm_pcon<<<KK / 16 / 8, 256, 0, stream>>>(pnewB, acc);
  palm_fin<<<1, 1, 0, stream>>>(acc, (float*)d_out);
}
